// TensorProductLayer_13950053777558
// MI455X (gfx1250) — matrix operand order repaired and checked by reading, not compiled, not run on hardware
//
#include <hip/hip_runtime.h>
#include <hip/hip_bf16.h>

// MI455X / gfx1250 fused TensorProductLayer.
// out = (G @ W2flat + f@B2) * (angular@Wa + ba),  G[n,k*64+i] = h[n,k]*f[n,i]
// Main GEMM: v_wmma_f32_16x16x32_f16 (A built in-registers by scaling cached
// f-fragments with h[m,k]); angular GEMM: v_wmma_f32_16x16x4_f32 (exact fp32,
// C-layout-compatible with main accumulators for the final elementwise mul).

typedef __attribute__((ext_vector_type(16))) _Float16 v16h;
typedef __attribute__((ext_vector_type(8)))  _Float16 v8h;
typedef __attribute__((ext_vector_type(8)))  float    v8f;
typedef __attribute__((ext_vector_type(2)))  float    v2f;

#define IN_F  64
#define OUT_F 64
#define RAD_F 32
#define ANG_F 16
#define HID   64
#define W2T_ELEMS (HID * IN_F * OUT_F)   // 262144 f16
#define B2T_ELEMS (IN_F * OUT_F)         // 4096 f16
#define SW2_PITCH 72                      // halves per o-row (pad vs 64 -> no bank conflicts)

// ---------------- prep: W2 -> f16 [k][o][i], b2 -> f16 [o][i] ----------------
__global__ void tpl_prep_kernel(const float* __restrict__ W2,
                                const float* __restrict__ b2,
                                _Float16* __restrict__ w2t,
                                _Float16* __restrict__ b2t) {
    int idx = blockIdx.x * 256 + threadIdx.x;
    if (idx < W2T_ELEMS) {
        int k = idx >> 12, rem = idx & 4095;
        int o = rem >> 6,  i   = rem & 63;
        w2t[idx] = (_Float16)W2[(k << 12) + (i << 6) + o];
    } else if (idx < W2T_ELEMS + B2T_ELEMS) {
        int e = idx - W2T_ELEMS;
        int o = e >> 6, i = e & 63;
        b2t[e] = (_Float16)b2[(i << 6) + o];
    }
}

__device__ inline v16h load16h(const _Float16* p) {
    v8h a = *(const v8h*)p;
    v8h b = *(const v8h*)(p + 8);
    v16h r;
#pragma unroll
    for (int t = 0; t < 8; ++t) { r[t] = a[t]; r[t + 8] = b[t]; }
    return r;
}

// ---------------- main fused kernel: 256 thr = 8 waves, 16 tokens/wave -------
__global__ void __launch_bounds__(256, 2)
tpl_main_kernel(const float* __restrict__ features,
                const float* __restrict__ radial,
                const float* __restrict__ angular,
                const float* __restrict__ W1,
                const float* __restrict__ b1,
                const float* __restrict__ Wa,
                const float* __restrict__ ba,
                const _Float16* __restrict__ w2t,
                const _Float16* __restrict__ b2t,
                float* __restrict__ out) {
    __shared__ float sW1[RAD_F * HID];            // 8 KB
    __shared__ float sB1[HID];                    // 256 B
    __shared__ float sWa[ANG_F * OUT_F];          // 4 KB
    __shared__ float sBa[OUT_F];                  // 256 B
    __shared__ float sH[8][HID * 16];             // 32 KB, [wave][k*16+m]
    __shared__ __align__(16) _Float16 sW2[2 * 64 * SW2_PITCH]; // 18 KB

    const int tid  = threadIdx.x;
    const int wave = tid >> 5;
    const int lane = tid & 31;
    const int hi   = lane >> 4;      // lane half (0/1)
    const int ln   = lane & 15;      // A-row m / C-col n
    const int tok0 = blockIdx.x * 128 + wave * 16;

    // ---- phase 0: stage small weights ----
    for (int u = tid; u < RAD_F * HID; u += 256) sW1[u] = W1[u];
    for (int u = tid; u < ANG_F * OUT_F; u += 256) sWa[u] = Wa[u];
    if (tid < HID)   sB1[tid] = b1[tid];
    if (tid < OUT_F) sBa[tid] = ba[tid];
    __syncthreads();

    // ---- phase 1a: h = silu(radial@W1+b1), stored transposed sH[k*16+m] ----
    {
        const float* rrow = radial + (size_t)(tok0 + ln) * RAD_F;
        float rad[RAD_F];
#pragma unroll
        for (int r = 0; r < RAD_F; r += 4) {
            float4 v = *(const float4*)(rrow + r);
            rad[r] = v.x; rad[r + 1] = v.y; rad[r + 2] = v.z; rad[r + 3] = v.w;
        }
        const int kbase = hi * 32;
        for (int j = 0; j < 32; ++j) {
            const int k = kbase + j;
            float acc = sB1[k];
#pragma unroll
            for (int r = 0; r < RAD_F; ++r) acc += rad[r] * sW1[r * HID + k];
            const float s = acc / (1.0f + __expf(-acc));   // silu
            sH[wave][k * 16 + ln] = s;
        }
    }

    // ---- phase 1b: cache f-fragments (A of 16x16x32 f16 WMMA, rows = tokens)
    // lane layout (contiguous k order chosen here; B matches): t=0..15 -> K=hi*16+t
    v16h Fa[2];
    {
        const float* frow = features + (size_t)(tok0 + ln) * IN_F;
#pragma unroll
        for (int c = 0; c < 2; ++c) {
            const int i0 = c * 32;
            float4 a0 = *(const float4*)(frow + i0 + hi * 16);
            float4 a1 = *(const float4*)(frow + i0 + hi * 16 + 4);
            float4 b0 = *(const float4*)(frow + i0 + hi * 16 + 8);
            float4 b1v = *(const float4*)(frow + i0 + hi * 16 + 12);
            Fa[c][0] = (_Float16)a0.x;  Fa[c][1] = (_Float16)a0.y;
            Fa[c][2] = (_Float16)a0.z;  Fa[c][3] = (_Float16)a0.w;
            Fa[c][4] = (_Float16)a1.x;  Fa[c][5] = (_Float16)a1.y;
            Fa[c][6] = (_Float16)a1.z;  Fa[c][7] = (_Float16)a1.w;
            Fa[c][8]  = (_Float16)b0.x; Fa[c][9]  = (_Float16)b0.y;
            Fa[c][10] = (_Float16)b0.z; Fa[c][11] = (_Float16)b0.w;
            Fa[c][12] = (_Float16)b1v.x; Fa[c][13] = (_Float16)b1v.y;
            Fa[c][14] = (_Float16)b1v.z; Fa[c][15] = (_Float16)b1v.w;
        }
    }

    // ---- phase 1c: init accumulators with bias term  f @ reshape(b2,[64,64])
    // B-frag (same contiguous k order as A): lane ln = col, t -> K = hi*16+t (contig in b2t [o][i])
    v8f C[4];
    {
        const v8f cz = {};
#pragma unroll
        for (int nt = 0; nt < 4; ++nt) {
            const _Float16* bp = b2t + (size_t)(nt * 16 + ln) * 64 + hi * 16;
            v16h B0 = load16h(bp);        // i0 = 0
            v16h B1 = load16h(bp + 32);   // i0 = 32
            C[nt] = __builtin_amdgcn_wmma_f32_16x16x32_f16(false, Fa[0], false, B0,
                                                           (short)0, cz, false, false);
            C[nt] = __builtin_amdgcn_wmma_f32_16x16x32_f16(false, Fa[1], false, B1,
                                                           (short)0, C[nt], false, false);
        }
    }

    // ---- phase 1d: ang = angular@Wa + ba via exact-fp32 WMMA 16x16x4 (K=16) --
    v8f Cang[4];
    {
#pragma unroll
        for (int nt = 0; nt < 4; ++nt) Cang[nt] = v8f{};
        const float* arow = angular + (size_t)(tok0 + ln) * ANG_F;
#pragma unroll
        for (int kc = 0; kc < 4; ++kc) {
            // A 16x4 f32: lanes 0-15 hold K=0,1; lanes 16-31 hold K=2,3
            float2 av = *(const float2*)(arow + kc * 4 + 2 * hi);
            v2f A; A[0] = av.x; A[1] = av.y;
#pragma unroll
            for (int nt = 0; nt < 4; ++nt) {
                v2f B;
                B[0] = sWa[(kc * 4 + 2 * hi + 0) * OUT_F + nt * 16 + ln];
                B[1] = sWa[(kc * 4 + 2 * hi + 1) * OUT_F + nt * 16 + ln];
                Cang[nt] = __builtin_amdgcn_wmma_f32_16x16x4_f32(false, A, false, B,
                                                                 (short)0, Cang[nt],
                                                                 false, false);
            }
        }
        const float bias0 = sBa[0 * 16 + ln], bias1 = sBa[1 * 16 + ln];
        const float bias2 = sBa[2 * 16 + ln], bias3 = sBa[3 * 16 + ln];
#pragma unroll
        for (int r = 0; r < 8; ++r) {
            Cang[0][r] += bias0; Cang[1][r] += bias1;
            Cang[2][r] += bias2; Cang[3][r] += bias3;
        }
    }

    // ---- main loop: out += G @ W2flat, W2 streamed through LDS 2 k-slices ----
    for (int kb = 0; kb < HID; kb += 2) {
        __syncthreads();   // previous chunk consumed
        // cooperative stage: 2 slices x 4096 halves, repack [o][i] pitch 64->72
        for (int u = tid; u < 1024; u += 256) {
            const int dk = u >> 9, s = u & 511;
            const int o = s >> 3, seg = s & 7;
            v8h d = *(const v8h*)(w2t + (size_t)(kb + dk) * 4096 + s * 8);
            *(v8h*)(sW2 + dk * 64 * SW2_PITCH + o * SW2_PITCH + seg * 8) = d;
        }
        __syncthreads();
#pragma unroll
        for (int dk = 0; dk < 2; ++dk) {
            const int k = kb + dk;
            const _Float16 hh = (_Float16)sH[wave][k * 16 + ln]; // h[m,k], broadcast read
            v16h A0, A1;
#pragma unroll
            for (int t = 0; t < 16; ++t) { A0[t] = Fa[0][t] * hh; A1[t] = Fa[1][t] * hh; }
            const _Float16* base = sW2 + dk * 64 * SW2_PITCH;
#pragma unroll
            for (int nt = 0; nt < 4; ++nt) {
                const _Float16* p = base + (nt * 16 + ln) * SW2_PITCH + hi * 16;
                v16h B0 = load16h(p);        // i0 = 0
                v16h B1 = load16h(p + 32);   // i0 = 32
                C[nt] = __builtin_amdgcn_wmma_f32_16x16x32_f16(false, A0, false, B0,
                                                               (short)0, C[nt], false, false);
                C[nt] = __builtin_amdgcn_wmma_f32_16x16x32_f16(false, A1, false, B1,
                                                               (short)0, C[nt], false, false);
            }
        }
    }

    // ---- epilogue: out = C * Cang ; C layout: m = r + 8*hi, n = ln ----------
#pragma unroll
    for (int nt = 0; nt < 4; ++nt) {
#pragma unroll
        for (int r = 0; r < 8; ++r) {
            const int mm = r + 8 * hi;
            out[(size_t)(tok0 + mm) * OUT_F + nt * 16 + ln] = C[nt][r] * Cang[nt][r];
        }
    }
}

extern "C" void kernel_launch(void* const* d_in, const int* in_sizes, int n_in,
                              void* d_out, int out_size, void* d_ws, size_t ws_size,
                              hipStream_t stream) {
    const float* features = (const float*)d_in[0];
    const float* radial   = (const float*)d_in[1];
    const float* angular  = (const float*)d_in[2];
    const float* W1 = (const float*)d_in[3];
    const float* b1 = (const float*)d_in[4];
    const float* W2 = (const float*)d_in[5];
    const float* b2 = (const float*)d_in[6];
    const float* Wa = (const float*)d_in[7];
    const float* ba = (const float*)d_in[8];
    float* out = (float*)d_out;

    _Float16* w2t = (_Float16*)d_ws;
    _Float16* b2t = w2t + W2T_ELEMS;

    const int N = in_sizes[0] / IN_F;           // 65536
    const int prep_blocks = (W2T_ELEMS + B2T_ELEMS + 255) / 256;
    tpl_prep_kernel<<<prep_blocks, 256, 0, stream>>>(W2, b2, w2t, b2t);
    tpl_main_kernel<<<N / 128, 256, 0, stream>>>(features, radial, angular,
                                                 W1, b1, Wa, ba, w2t, b2t, out);
}
